// MusicLSTM_32186484917015
// MI455X (gfx1250) — compile-verified, matrix operand order checked by reading, not run on hardware
//
#include <hip/hip_runtime.h>
#include <hip/hip_bf16.h>

// ---------------------------------------------------------------------------
// MusicLSTM for MI455X (gfx1250): bf16 WMMA GEMMs (register-tiled) +
// persistent-WGP LSTM scans. B=32, T=1024. Intermediates (~180MB) fit the
// 192MB L2 -> compute/latency bound, so maximize WMMA utilization and
// FLOP/byte out of L2, and keep the serial recurrences spill-free and
// branch-free on one WGP.
// ---------------------------------------------------------------------------

typedef __bf16 bf16;
typedef __attribute__((ext_vector_type(8)))  bf16  v8bf;
typedef __attribute__((ext_vector_type(16))) bf16  v16bf;
typedef __attribute__((ext_vector_type(8)))  float v8f;

#define BB 32          // batch
#define TT 1024        // timesteps
#define BT (BB * TT)   // 32768 rows

// Build one WMMA A/B fragment (16 bf16) for this lane from a row-major
// [rows, K] matrix. Per the CDNA5 16-bit A/B layout, lane (lane16, hi) holds
// K = {hi*8 .. hi*8+7} and {16+hi*8 .. 16+hi*8+7}: two contiguous 16B runs.
// `base` must already point at  row*K + k0 + hi*8.
__device__ __forceinline__ v16bf load_frag(const bf16* base) {
  v8bf lo = *(const v8bf*)(base);
  v8bf hh = *(const v8bf*)(base + 16);
  return __builtin_shufflevector(lo, hh, 0, 1, 2, 3, 4, 5, 6, 7,
                                         8, 9, 10, 11, 12, 13, 14, 15);
}

__device__ __forceinline__ v8f wmma_bf16(v16bf a, v16bf b, v8f c) {
  return __builtin_amdgcn_wmma_f32_16x16x32_bf16(false, a, false, b,
                                                 (short)0, c, false, false);
}

// Branch-free activations: v_exp_f32 + v_rcp_f32 only (no exec-mask paths
// in the serial recurrence). Saturate correctly for large |x|.
__device__ __forceinline__ float sigmoidf_(float x) {
  return __builtin_amdgcn_rcpf(1.0f + __expf(-x));
}
__device__ __forceinline__ float tanhf_(float x) {
  return 2.0f * __builtin_amdgcn_rcpf(1.0f + __expf(-2.0f * x)) - 1.0f;
}

// ---------------------------------------------------------------------------
// Elementwise f32 -> bf16 weight conversion.
// ---------------------------------------------------------------------------
__global__ void cvt_f32_bf16(const float* __restrict__ s, bf16* __restrict__ d,
                             int n) {
  int i = blockIdx.x * blockDim.x + threadIdx.x;
  if (i < n) d[i] = (bf16)s[i];
}

// ---------------------------------------------------------------------------
// Register-tiled GEMM:
//   O[M,N] = A[M,K](bf16,row-major) * W[N,K](bf16,row-major)^T + bias0+bias1
// Each wave computes a (16*TM) x (16*TN) register tile: per K-step it loads
// TM+TN fragments and issues TM*TN WMMAs (4x4 -> 16 WMMA / 16 b128 loads,
// 4x the FLOP/byte of an untiled wave). K%32==0.
// ---------------------------------------------------------------------------
template <int TM, int TN>
__global__ void gemm_bf16_wmma(const bf16* __restrict__ A,
                               const bf16* __restrict__ W,
                               const float* __restrict__ b0,
                               const float* __restrict__ b1,
                               bf16* __restrict__ O,
                               int M, int N, int K) {
  const int wg = blockIdx.x * (blockDim.x >> 5) + (threadIdx.x >> 5);
  const int ntiles = N / (16 * TN);
  const int mtile = wg / ntiles;
  const int ntile = wg - mtile * ntiles;
  if (mtile >= M / (16 * TM)) return;  // wave-uniform

  const int lane = threadIdx.x & 31;
  const int lane16 = lane & 15;
  const int hi = lane >> 4;

  const bf16* arow[TM];
  const bf16* wrow[TN];
#pragma unroll
  for (int i = 0; i < TM; ++i)
    arow[i] = A + (size_t)(mtile * 16 * TM + i * 16 + lane16) * K + hi * 8;
#pragma unroll
  for (int j = 0; j < TN; ++j)
    wrow[j] = W + (size_t)(ntile * 16 * TN + j * 16 + lane16) * K + hi * 8;

  v8f acc[TM][TN];
#pragma unroll
  for (int i = 0; i < TM; ++i)
#pragma unroll
    for (int j = 0; j < TN; ++j) acc[i][j] = (v8f){};

  for (int k = 0; k < K; k += 32) {
    v16bf af[TM], wf[TN];
#pragma unroll
    for (int i = 0; i < TM; ++i) af[i] = load_frag(arow[i] + k);
#pragma unroll
    for (int j = 0; j < TN; ++j) wf[j] = load_frag(wrow[j] + k);
#pragma unroll
    for (int i = 0; i < TM; ++i)
#pragma unroll
      for (int j = 0; j < TN; ++j)
        acc[i][j] = wmma_bf16(af[i], wf[j], acc[i][j]);
  }

#pragma unroll
  for (int j = 0; j < TN; ++j) {
    const int n = ntile * 16 * TN + j * 16 + lane16;
    const float bias = (b0 ? b0[n] : 0.0f) + (b1 ? b1[n] : 0.0f);
#pragma unroll
    for (int i = 0; i < TM; ++i) {
#pragma unroll
      for (int r = 0; r < 8; ++r) {   // CDNA5 C layout: M = r + hi*8
        const int m = mtile * 16 * TM + i * 16 + hi * 8 + r;
        O[(size_t)m * N + n] = (bf16)(acc[i][j][r] + bias);
      }
    }
  }
}

// ---------------------------------------------------------------------------
// Persistent single-workgroup LSTM scan, H in {64, 256}.
//   xproj : [BT, 4H] bf16 (precomputed x @ w_ih^T + b_ih + b_hh)
//   whh   : [4H, H]  bf16 (PyTorch layout, gate order i,f,g,o)
//   hout  : [BT, H]  bf16
// One wave per unit-tile (jtile) owns BOTH batch tiles and all four gates:
// every w_hh fragment is loaded once per step and feeds two WMMAs, so L2
// weight traffic hits its 512KB/step minimum (H=256). For H=64 the entire
// w_hh (32KB) is staged in LDS once, so the recurrent step touches only LDS
// (weights + h) plus xproj/hout. h is double-buffered bf16 in LDS, c is f32
// in LDS; one barrier per timestep. blockDim = 2*H (H/16 waves), single
// workgroup -> __launch_bounds__(.,1) gives the full register budget
// (no spills in the serial hot loop).
// ---------------------------------------------------------------------------
template <int H>
__global__ __launch_bounds__(2 * H, 1)
void lstm_scan_wmma(const bf16* __restrict__ xproj,
                    const bf16* __restrict__ whh,
                    bf16* __restrict__ hout) {
  constexpr int KT = H / 32;        // K steps
  constexpr bool WLDS = (H == 64);  // 4H*H bf16 = 32KB fits LDS (H=64 only)
  __shared__ bf16 hbuf[2][BB][H];
  __shared__ float cbuf[BB][H];
  __shared__ bf16 wlds[WLDS ? 4 * H * H : 16];

  const int tid = threadIdx.x;
  for (int i = tid; i < BB * H; i += blockDim.x) {
    ((bf16*)hbuf)[i] = (bf16)0.0f;  // hbuf[0] = 0
    ((float*)cbuf)[i] = 0.0f;
  }
  if constexpr (WLDS) {  // stage w_hh into LDS (b128 copies)
    const int4* src = (const int4*)whh;
    int4* dst = (int4*)wlds;
    for (int i = tid; i < 4 * H * H * 2 / 16; i += blockDim.x) dst[i] = src[i];
  }
  __syncthreads();

  const int jtile = tid >> 5;  // one wave per unit tile
  const int lane = tid & 31;
  const int lane16 = lane & 15;
  const int hi = lane >> 4;
  const int j = jtile * 16 + lane16;  // output unit column

  const bf16* wbase[4];
#pragma unroll
  for (int g = 0; g < 4; ++g) {
    if constexpr (WLDS)
      wbase[g] = wlds + (size_t)(g * H + j) * H + hi * 8;  // ds_load path
    else
      wbase[g] = whh + (size_t)(g * H + j) * H + hi * 8;   // L2 path
  }

  for (int t = 0; t < TT; ++t) {
    const bf16* hcur = &hbuf[t & 1][0][0];
    v8f acc[2][4];
#pragma unroll
    for (int mt = 0; mt < 2; ++mt)
#pragma unroll
      for (int g = 0; g < 4; ++g) acc[mt][g] = (v8f){};

#pragma unroll
    for (int k = 0; k < KT; ++k) {
      v16bf af0 = load_frag(hcur + (0 * 16 + lane16) * H + k * 32 + hi * 8);
      v16bf af1 = load_frag(hcur + (1 * 16 + lane16) * H + k * 32 + hi * 8);
#pragma unroll
      for (int g = 0; g < 4; ++g) {
        v16bf wf = load_frag(wbase[g] + k * 32);
        acc[0][g] = wmma_bf16(af0, wf, acc[0][g]);
        acc[1][g] = wmma_bf16(af1, wf, acc[1][g]);
      }
    }

#pragma unroll
    for (int mt = 0; mt < 2; ++mt) {
#pragma unroll
      for (int r = 0; r < 8; ++r) {
        const int m = mt * 16 + hi * 8 + r;  // batch index
        const size_t row = (size_t)m * TT + t;
        const bf16* xp = xproj + row * (4 * H);
        float gi = acc[mt][0][r] + (float)xp[0 * H + j];
        float gf = acc[mt][1][r] + (float)xp[1 * H + j];
        float gg = acc[mt][2][r] + (float)xp[2 * H + j];
        float go = acc[mt][3][r] + (float)xp[3 * H + j];
        gi = sigmoidf_(gi);
        gf = sigmoidf_(gf);
        go = sigmoidf_(go);
        gg = tanhf_(gg);
        float c = gf * cbuf[m][j] + gi * gg;
        cbuf[m][j] = c;
        float h = go * tanhf_(c);
        hbuf[(t + 1) & 1][m][j] = (bf16)h;
        hout[row * H + j] = (bf16)h;
      }
    }
    if (t + 1 < TT) {  // pre-touch next timestep's gate rows (global_prefetch_b8)
      __builtin_prefetch(xproj + ((size_t)(hi * 8) * TT + t + 1) * (4 * H) + j, 0, 1);
      __builtin_prefetch(xproj + ((size_t)(16 + hi * 8) * TT + t + 1) * (4 * H) + j, 0, 1);
    }
    __syncthreads();
  }
}

// ---------------------------------------------------------------------------
// K=15 input projection (too narrow for WMMA): out[BT,256] bf16.
// ---------------------------------------------------------------------------
__global__ void xproj_in15(const float* __restrict__ x,
                           const float* __restrict__ w,
                           const float* __restrict__ b0,
                           const float* __restrict__ b1,
                           bf16* __restrict__ out) {
  const int idx = blockIdx.x * blockDim.x + threadIdx.x;  // BT*256 threads
  const int row = idx >> 8;
  const int g = idx & 255;
  const float* xr = x + row * 15;
  const float* wr = w + g * 15;
  float s = b0[g] + b1[g];
#pragma unroll
  for (int i = 0; i < 15; ++i) s += xr[i] * wr[i];
  out[(size_t)row * 256 + g] = (bf16)s;
}

// ---------------------------------------------------------------------------
// Final FC 256 -> 15 (N=15 not a tile multiple): f32 output.
// ---------------------------------------------------------------------------
__global__ void fc_out(const bf16* __restrict__ h, const float* __restrict__ w,
                       const float* __restrict__ b, float* __restrict__ out) {
  const int idx = blockIdx.x * blockDim.x + threadIdx.x;
  if (idx >= BT * 15) return;
  const int row = idx / 15;
  const int n = idx - row * 15;
  const bf16* hr = h + (size_t)row * 256;
  const float* wr = w + n * 256;
  float s = b[n];
  for (int k = 0; k < 256; ++k) s += (float)hr[k] * wr[k];
  out[idx] = s;
}

// ---------------------------------------------------------------------------
// Host orchestration.
// ---------------------------------------------------------------------------
template <int TM, int TN>
static inline int gemm_blocks(int M, int N) {
  return (M / (16 * TM)) * (N / (16 * TN)) / 8;  // 8 waves per 256-thread block
}

extern "C" void kernel_launch(void* const* d_in, const int* in_sizes, int n_in,
                              void* d_out, int out_size, void* d_ws,
                              size_t ws_size, hipStream_t stream) {
  (void)in_sizes; (void)n_in; (void)out_size; (void)ws_size;
  const float* x       = (const float*)d_in[0];
  const float* w_ih1_0 = (const float*)d_in[1];
  const float* w_hh1_0 = (const float*)d_in[2];
  const float* b_ih1_0 = (const float*)d_in[3];
  const float* b_hh1_0 = (const float*)d_in[4];
  const float* w_ih1_1 = (const float*)d_in[5];
  const float* w_hh1_1 = (const float*)d_in[6];
  const float* b_ih1_1 = (const float*)d_in[7];
  const float* b_hh1_1 = (const float*)d_in[8];
  const float* dense_w = (const float*)d_in[9];
  const float* dense_b = (const float*)d_in[10];
  const float* w_ih2_0 = (const float*)d_in[11];
  const float* w_hh2_0 = (const float*)d_in[12];
  const float* b_ih2_0 = (const float*)d_in[13];
  const float* b_hh2_0 = (const float*)d_in[14];
  const float* w_ih2_1 = (const float*)d_in[15];
  const float* w_hh2_1 = (const float*)d_in[16];
  const float* b_ih2_1 = (const float*)d_in[17];
  const float* b_hh2_1 = (const float*)d_in[18];
  const float* fc_w    = (const float*)d_in[19];
  const float* fc_b    = (const float*)d_in[20];

  // Workspace carve-up (bump allocator, 256B aligned). Peak ~180 MB.
  char* ws = (char*)d_ws;
  size_t off = 0;
  auto alloc = [&](size_t bytes) -> void* {
    void* p = ws + off;
    off = (off + bytes + 255) & ~(size_t)255;
    return p;
  };
  bf16* xprojbuf  = (bf16*)alloc((size_t)BT * 1024 * 2);  // reused by all 4 layers
  bf16* dense_act = (bf16*)alloc((size_t)BT * 1024 * 2);
  bf16* hA = (bf16*)alloc((size_t)BT * 64 * 2);
  bf16* hB = (bf16*)alloc((size_t)BT * 64 * 2);
  bf16* hC = (bf16*)alloc((size_t)BT * 256 * 2);
  bf16* hD = (bf16*)alloc((size_t)BT * 256 * 2);
  bf16* wb_hh1_0 = (bf16*)alloc(256 * 64 * 2);
  bf16* wb_ih1_1 = (bf16*)alloc(256 * 64 * 2);
  bf16* wb_hh1_1 = (bf16*)alloc(256 * 64 * 2);
  bf16* wb_dense = (bf16*)alloc(1024 * 64 * 2);
  bf16* wb_ih2_0 = (bf16*)alloc((size_t)1024 * 1024 * 2);
  bf16* wb_hh2_0 = (bf16*)alloc((size_t)1024 * 256 * 2);
  bf16* wb_ih2_1 = (bf16*)alloc((size_t)1024 * 256 * 2);
  bf16* wb_hh2_1 = (bf16*)alloc((size_t)1024 * 256 * 2);

  // 0) Weight conversions to bf16.
  struct { const float* s; bf16* d; int n; } cv[] = {
      {w_hh1_0, wb_hh1_0, 256 * 64},   {w_ih1_1, wb_ih1_1, 256 * 64},
      {w_hh1_1, wb_hh1_1, 256 * 64},   {dense_w, wb_dense, 1024 * 64},
      {w_ih2_0, wb_ih2_0, 1024 * 1024},{w_hh2_0, wb_hh2_0, 1024 * 256},
      {w_ih2_1, wb_ih2_1, 1024 * 256}, {w_hh2_1, wb_hh2_1, 1024 * 256},
  };
  for (auto& c : cv)
    cvt_f32_bf16<<<(c.n + 255) / 256, 256, 0, stream>>>(c.s, c.d, c.n);

  // 1) x_projA = x @ w_ih1_0^T + b  (K=15, scalar kernel)
  xproj_in15<<<BT, 256, 0, stream>>>(x, w_ih1_0, b_ih1_0, b_hh1_0, xprojbuf);
  // 2) scan layer 1_0 (H=64): 4 waves, w_hh in LDS
  lstm_scan_wmma<64><<<1, 128, 0, stream>>>(xprojbuf, wb_hh1_0, hA);
  // 3) x_projB = hA @ w_ih1_1^T + b   (K=64 -> 2x2 tile)
  gemm_bf16_wmma<2, 2><<<gemm_blocks<2, 2>(BT, 256), 256, 0, stream>>>(
      hA, wb_ih1_1, b_ih1_1, b_hh1_1, xprojbuf, BT, 256, 64);
  // 4) scan layer 1_1 (H=64)
  lstm_scan_wmma<64><<<1, 128, 0, stream>>>(xprojbuf, wb_hh1_1, hB);
  // 5) dense: hB @ dense_w^T + dense_b   (K=64 -> 2x2 tile)
  gemm_bf16_wmma<2, 2><<<gemm_blocks<2, 2>(BT, 1024), 256, 0, stream>>>(
      hB, wb_dense, dense_b, nullptr, dense_act, BT, 1024, 64);
  // 6) x_projC = dense @ w_ih2_0^T + b   (69 GF, K=1024 -> 4x4 tile)
  gemm_bf16_wmma<4, 4><<<gemm_blocks<4, 4>(BT, 1024), 256, 0, stream>>>(
      dense_act, wb_ih2_0, b_ih2_0, b_hh2_0, xprojbuf, BT, 1024, 1024);
  // 7) scan layer 2_0 (H=256): 16 waves, w_hh read once per step from L2
  lstm_scan_wmma<256><<<1, 512, 0, stream>>>(xprojbuf, wb_hh2_0, hC);
  // 8) x_projD = hC @ w_ih2_1^T + b   (K=256 -> 4x4 tile)
  gemm_bf16_wmma<4, 4><<<gemm_blocks<4, 4>(BT, 1024), 256, 0, stream>>>(
      hC, wb_ih2_1, b_ih2_1, b_hh2_1, xprojbuf, BT, 1024, 256);
  // 9) scan layer 2_1 (H=256)
  lstm_scan_wmma<256><<<1, 512, 0, stream>>>(xprojbuf, wb_hh2_1, hD);
  // 10) final FC 256 -> 15 (f32 out)
  fc_out<<<(BT * 15 + 255) / 256, 256, 0, stream>>>(hD, fc_w, fc_b,
                                                    (float*)d_out);
}
